// KeypointDetector_35296041238937
// MI455X (gfx1250) — matrix operand order repaired and checked by reading, not compiled, not run on hardware
//
#include <hip/hip_runtime.h>
#include <stdint.h>

// ---------------- problem constants (from setup_inputs) ----------------
#define H_IMG 1024
#define W_IMG 1024
#define BATCH 16
#define CHN   3
#define KSEL  512
#define MARGIN 5
#define K_HARRIS 0.04f

// ---------------- vector types ----------------
typedef _Float16 v16h __attribute__((ext_vector_type(16)));
typedef float    v8f  __attribute__((ext_vector_type(8)));
typedef unsigned int u32x4 __attribute__((ext_vector_type(4)));
typedef int          i32x8 __attribute__((ext_vector_type(8)));
typedef int          i32x4 __attribute__((ext_vector_type(4)));

// ---------------- workspace layout ----------------
#define OFF_HIST  ((size_t)BATCH * H_IMG * W_IMG * 4)
#define OFF_STATE (OFF_HIST + (size_t)BATCH * 256 * 4)
#define OFF_CAND  (OFF_STATE + 192)
#define CAND_CAP  4096

// sHT geometry: [3 mats][36 cols][56 rows] of f16, zero padded.
//   column index = H-col + 2   (H-col in [-2,33])
//   row index    = H-row + 2   (H-row in [-2,53] read window, valid data 0..33)
#define HT_PITCH 56
#define HT_COLS  36
#define HT_MAT   (HT_COLS * HT_PITCH)   // 2016 halves per matrix

// =====================================================================
// Kernel 1: fused Harris response. One block = 32x32 output tile.
//  - 36x36x3 halo region DMA'd into LDS by the Tensor Data Mover
//  - vertical Gaussian as v_wmma_f32_16x16x32_f16 per 16x16 subtile
//    (A = constant banded {1,2,1}/16, K zero-padded to 32)
// =====================================================================
__global__ __launch_bounds__(128) void harris_fused(const float* __restrict__ x,
                                                    float* __restrict__ R)
{
    __shared__ float    sC[3 * 36 * 36];     // 3 channel tiles; slab0 becomes gray
    __shared__ float    sP[3 * 34 * 34];     // dx*dx, dy*dy, dx*dy
    __shared__ _Float16 sHT[3 * HT_MAT];     // transposed, zero-padded smoothed products

    const int bx = blockIdx.x, by = blockIdx.y, b = blockIdx.z;
    const int tid = threadIdx.x;

    // clamped load origin (shifted halo only affects masked margin pixels)
    int gx0 = 32 * bx - 2; gx0 = gx0 < 0 ? 0 : (gx0 > W_IMG - 36 ? W_IMG - 36 : gx0);
    int gy0 = 32 * by - 2; gy0 = gy0 < 0 ? 0 : (gy0 > H_IMG - 36 ? H_IMG - 36 : gy0);
    const int ox = 32 * bx - gx0;   // 0, 2 or 4
    const int oy = 32 * by - gy0;

    // ---- TDM: one 3-D tile load (36 x 36 x 3ch) into LDS, issued by wave 0 ----
    if (tid < 32) {
        unsigned long long gaddr = (unsigned long long)(uintptr_t)x +
            4ull * ((unsigned long long)b * CHN * H_IMG * W_IMG +
                    (unsigned long long)gy0 * W_IMG + (unsigned long long)gx0);
        unsigned int lds = (unsigned int)(uintptr_t)&sC[0];

        u32x4 g0; i32x8 g1; i32x4 g2, g3;
        // group0: count=1 | lds_addr | global_addr[56:0] | type=2
        g0[0] = 1u;
        g0[1] = lds;
        g0[2] = (unsigned int)(gaddr & 0xffffffffu);
        g0[3] = (unsigned int)((gaddr >> 32) & 0x1ffffffu) | 0x80000000u;
        // group1: data_size=4B; tensor 1024x1024(x3); tile 36x36x3;
        //         dim0_stride=1024; dim1_stride=1024*1024 (= channel stride)
        g1[0] = (int)(2u << 16);                          // data_size code 2 -> 4 bytes
        g1[1] = (int)((unsigned)W_IMG << 16);             // tensor_dim0[15:0]
        g1[2] = (int)((unsigned)H_IMG << 16);             // dim0 hi=0 | tensor_dim1 lo
        g1[3] = (int)(36u << 16);                         // dim1 hi=0 | tile_dim0=36
        g1[4] = (int)(36u | (3u << 16));                  // tile_dim1=36 | tile_dim2=3
        g1[5] = (int)(unsigned)W_IMG;                     // tensor_dim0_stride lo32
        g1[6] = 0;                                        // stride0 hi16 | stride1[15:0]=0
        g1[7] = (int)(((unsigned)(H_IMG * W_IMG)) >> 16); // stride1[47:16]
        g2[0] = CHN; g2[1] = 0; g2[2] = 0; g2[3] = 0;     // tensor_dim2=3
        g3[0] = 0; g3[1] = 0; g3[2] = 0; g3[3] = 0;
#if __clang_major__ >= 23
        i32x8 g4 = {0, 0, 0, 0, 0, 0, 0, 0};
        __builtin_amdgcn_tensor_load_to_lds(g0, g1, g2, g3, g4, 0);
#else
        __builtin_amdgcn_tensor_load_to_lds(g0, g1, g2, g3, 0);
#endif
    }

    // ---- zero the padded transpose buffer while the DMA is in flight ----
    {
        unsigned int* z = (unsigned int*)sHT;
        for (int i = tid; i < (3 * HT_MAT) / 2; i += 128) z[i] = 0u;
    }
    if (tid < 32) __builtin_amdgcn_s_wait_tensorcnt(0);
    __syncthreads();

    // ---- grayscale mean over channels, in place into slab 0 ----
    for (int i = tid; i < 36 * 36; i += 128) {
        sC[i] = (sC[i] + sC[1296 + i] + sC[2592 + i]) * (1.0f / 3.0f);
    }
    __syncthreads();

    // ---- derivatives + structure-tensor products over 34x34 ----
    for (int i = tid; i < 34 * 34; i += 128) {
        int py = i / 34, px = i % 34;
        int y = py + 1, xg = px + 1;
        const float* G = sC;
        float a00 = G[(y - 1) * 36 + xg - 1], a01 = G[(y - 1) * 36 + xg], a02 = G[(y - 1) * 36 + xg + 1];
        float a10 = G[y * 36 + xg - 1],                                   a12 = G[y * 36 + xg + 1];
        float a20 = G[(y + 1) * 36 + xg - 1], a21 = G[(y + 1) * 36 + xg], a22 = G[(y + 1) * 36 + xg + 1];
        float dx = (a02 + a12 + a22) - (a00 + a10 + a20);
        float dy = (a20 + a21 + a22) - (a00 + a01 + a02);
        sP[i] = dx * dx; sP[1156 + i] = dy * dy; sP[2312 + i] = dx * dy;
    }
    __syncthreads();

    // ---- horizontal {1,2,1} smoothing -> f16, stored TRANSPOSED+padded ----
    for (int i = tid; i < 34 * 32; i += 128) {
        int r = i >> 5, c = i & 31;
#pragma unroll
        for (int m = 0; m < 3; ++m) {
            const float* P = sP + m * 1156 + r * 34;
            float h = P[c] + 2.0f * P[c + 1] + P[c + 2];
            sHT[m * HT_MAT + (c + 2) * HT_PITCH + (r + 2)] = (_Float16)h;
        }
    }
    __syncthreads();

    // ---- vertical {1,2,1}/16 via WMMA: one 16x16 subtile per wave ----
    const int wave = tid >> 5;
    const int lane = tid & 31;
    const int sm = (wave >> 1) << 4;
    const int sn = (wave & 1) << 4;
    const int N  = lane & 15;
    const int hiW = lane >> 4;

    // A (16x32 f16, constant banded weights). Contiguous k order (same as B):
    // lanes 0-15: M=lane, K=0..15; lanes 16-31: M=lane-16, K=16..31
    v16h Av;
#pragma unroll
    for (int s = 0; s < 16; ++s) {
        int K = s + (hiW ? 16 : 0);
        int d = K - N;
        float w = (d == 1) ? 0.125f : ((d == 0 || d == 2) ? 0.0625f : 0.0f);
        Av[s] = (_Float16)w;
    }

    // B (32x16 f16): lanes 0-15 hold K=0..15, lanes 16-31 hold K=16..31, N=lane&15.
    // K runs along the contiguous LDS axis -> unconditional vectorizable loads.
    // Out-of-band K entries read zero padding; A's banded zeros make them inert.
    const int r0 = sm + oy - 2;                                // base H row (K=0 tap)
    const int nc = sn + N + ox - 2;                            // H column, in [-2,33]
    const int baseNH = (nc + 2) * HT_PITCH + (r0 + 2) + (hiW ? 16 : 0);

    v8f S[3];
#pragma unroll
    for (int m = 0; m < 3; ++m) {
        const _Float16* src = sHT + m * HT_MAT + baseNH;
        v16h Bv;
#pragma unroll
        for (int s = 0; s < 16; ++s) Bv[s] = src[s];
        v8f c = {0.f, 0.f, 0.f, 0.f, 0.f, 0.f, 0.f, 0.f};
        S[m] = __builtin_amdgcn_wmma_f32_16x16x32_f16(false, Av, false, Bv,
                                                      (short)0, c, false, false);
    }

    // ---- Harris response, mask, store. D layout: lane<16 -> M=j, else M=8+j ----
    const int hbase = 32 * by + sm + (hiW ? 8 : 0);
    const int wcol  = 32 * bx + sn + N;
    const bool colMask = (wcol >= MARGIN) && (wcol < W_IMG - MARGIN);
    float* Rb = R + (size_t)b * H_IMG * W_IMG;
#pragma unroll
    for (int j = 0; j < 8; ++j) {
        int h = hbase + j;
        float xx = S[0][j], yy = S[1][j], xy = S[2][j];
        float det = xx * yy - xy * xy;
        float tr  = xx + yy;
        float r   = det - K_HARRIS * tr * tr;
        bool mask = colMask && (h >= MARGIN) && (h < H_IMG - MARGIN);
        Rb[h * W_IMG + wcol] = mask ? r : 0.0f;
    }
}

// =====================================================================
// Top-k selection: byte-radix select on order keys, then gather+sort.
// =====================================================================
__device__ __forceinline__ unsigned int orderKey(float f)
{
    unsigned int u = __float_as_uint(f);
    return (u & 0x80000000u) ? ~u : (u | 0x80000000u);
}

__global__ __launch_bounds__(256) void select_init(unsigned int* __restrict__ hist,
                                                   unsigned int* __restrict__ state)
{
    for (int j = threadIdx.x; j < BATCH * 256; j += 256) hist[j] = 0u;
    if (threadIdx.x < BATCH) {
        state[threadIdx.x]      = 0u;    // prefix
        state[16 + threadIdx.x] = KSEL;  // remaining k
        state[32 + threadIdx.x] = 0u;    // candidate count
    }
}

__global__ __launch_bounds__(256) void select_hist(const float* __restrict__ R,
                                                   unsigned int* __restrict__ hist,
                                                   const unsigned int* __restrict__ state,
                                                   int shift)
{
    __shared__ unsigned int lh[256];
    const int b = blockIdx.y;
    lh[threadIdx.x] = 0u;
    __syncthreads();
    const unsigned int pfx = state[b];
    const float* Rb = R + (size_t)b * (H_IMG * W_IMG);
    const int base = blockIdx.x * 16384;
    for (int i = threadIdx.x; i < 16384; i += 256) {
        unsigned int key = orderKey(Rb[base + i]);
        unsigned int hi = (shift == 24) ? 0u : (key >> (shift + 8));
        if (hi == pfx) atomicAdd(&lh[(key >> shift) & 255u], 1u);
    }
    __syncthreads();
    unsigned int v = lh[threadIdx.x];
    if (v) atomicAdd(&hist[b * 256 + threadIdx.x], v);
}

__global__ __launch_bounds__(256) void select_scan(unsigned int* __restrict__ hist,
                                                   unsigned int* __restrict__ state)
{
    const int b = blockIdx.x;
    if (threadIdx.x == 0) {
        unsigned int remK = state[16 + b];
        unsigned int pfx  = state[b];
        unsigned int cum = 0u;
        int sel = 0;
        for (int bin = 255; bin >= 0; --bin) {
            unsigned int c = hist[b * 256 + bin];
            if (cum + c >= remK) { sel = bin; break; }
            cum += c;
        }
        state[b]      = (pfx << 8) | (unsigned int)sel;
        state[16 + b] = remK - cum;
    }
    __syncthreads();
    hist[b * 256 + threadIdx.x] = 0u;   // ready for next pass
}

__global__ __launch_bounds__(256) void select_gather(const float* __restrict__ R,
                                                     const unsigned int* __restrict__ state,
                                                     unsigned int* __restrict__ cnt,
                                                     unsigned long long* __restrict__ cand)
{
    const int b = blockIdx.y;
    const unsigned int T = state[b];          // exact k-th largest key
    const float* Rb = R + (size_t)b * (H_IMG * W_IMG);
    const int base = blockIdx.x * 16384;
    for (int i = threadIdx.x; i < 16384; i += 256) {
        int idx = base + i;
        unsigned int key = orderKey(Rb[idx]);
        if (key >= T) {
            unsigned int pos = atomicAdd(&cnt[b], 1u);
            if (pos < CAND_CAP)
                cand[(size_t)b * CAND_CAP + pos] =
                    ((unsigned long long)key << 20) |
                    (unsigned long long)(0xFFFFFu - (unsigned int)idx);
        }
    }
}

__global__ __launch_bounds__(512) void select_emit(const unsigned long long* __restrict__ cand,
                                                   const unsigned int* __restrict__ cnt,
                                                   float* __restrict__ out)
{
    __shared__ unsigned long long s[CAND_CAP];
    const int b = blockIdx.x;
    unsigned int n = cnt[b]; if (n > CAND_CAP) n = CAND_CAP;
    for (int i = threadIdx.x; i < CAND_CAP; i += 512)
        s[i] = (i < (int)n) ? cand[(size_t)b * CAND_CAP + i] : 0ull;
    __syncthreads();

    // bitonic sort, descending composite (= value desc, index asc on ties)
    for (int k = 2; k <= CAND_CAP; k <<= 1) {
        for (int j = k >> 1; j > 0; j >>= 1) {
            for (int i = threadIdx.x; i < CAND_CAP; i += 512) {
                int ixj = i ^ j;
                if (ixj > i) {
                    unsigned long long a = s[i], c = s[ixj];
                    bool up = ((i & k) == 0);
                    if (up ? (a < c) : (a > c)) { s[i] = c; s[ixj] = a; }
                }
            }
            __syncthreads();
        }
    }

    if (threadIdx.x < KSEL) {
        unsigned long long comp = s[threadIdx.x];
        unsigned int idx = 0xFFFFFu - (unsigned int)(comp & 0xFFFFFu);
        int row = (int)(idx >> 10), col = (int)(idx & 1023u);
        float cy = ((float)row * (1.0f / (float)H_IMG) - 0.5f) * 2.0f;
        float cx = ((float)col * (1.0f / (float)W_IMG) - 0.5f) * 2.0f;
        cy = fminf(1.0f, fmaxf(-1.0f, cy));
        cx = fminf(1.0f, fmaxf(-1.0f, cx));
        out[((size_t)b * KSEL + threadIdx.x) * 2 + 0] = cy;
        out[((size_t)b * KSEL + threadIdx.x) * 2 + 1] = cx;
    }
}

// =====================================================================
extern "C" void kernel_launch(void* const* d_in, const int* in_sizes, int n_in,
                              void* d_out, int out_size, void* d_ws, size_t ws_size,
                              hipStream_t stream)
{
    (void)in_sizes; (void)n_in; (void)out_size; (void)ws_size;
    const float* x = (const float*)d_in[0];         // [16,3,1024,1024] f32
    float* out = (float*)d_out;                     // [16,512,2] f32

    float* R                 = (float*)d_ws;
    unsigned int* hist       = (unsigned int*)((char*)d_ws + OFF_HIST);
    unsigned int* state      = (unsigned int*)((char*)d_ws + OFF_STATE);
    unsigned int* cnt        = state + 32;
    unsigned long long* cand = (unsigned long long*)((char*)d_ws + OFF_CAND);

    harris_fused<<<dim3(32, 32, BATCH), 128, 0, stream>>>(x, R);
    select_init<<<1, 256, 0, stream>>>(hist, state);
    for (int p = 0; p < 4; ++p) {
        int shift = 24 - 8 * p;
        select_hist<<<dim3(64, BATCH), 256, 0, stream>>>(R, hist, state, shift);
        select_scan<<<BATCH, 256, 0, stream>>>(hist, state);
    }
    select_gather<<<dim3(64, BATCH), 256, 0, stream>>>(R, state, cnt, cand);
    select_emit<<<BATCH, 512, 0, stream>>>(cand, cnt, out);
}
